// GraphSAGENet_15367392985610
// MI455X (gfx1250) — matrix operand order repaired and checked by reading, not compiled, not run on hardware
//
#include <hip/hip_runtime.h>

// ---------------------------------------------------------------------------
// GraphSAGE (2-layer, mean aggr) for MI455X / gfx1250, wave32 + WMMA.
//   layer: h = relu( (segmean_{dst} x[src]) @ w_l + b + x @ w_r )
//   out   = h2 @ w_lin + b_lin
// Aggregation = float atomics (memory bound, L2 resident).
// Dense 96x96 linears = v_wmma_f32_16x16x32_f16 tiles, f32 accumulation,
// with TWO independent accumulator chains to hide WMMA data hazards.
// ---------------------------------------------------------------------------

typedef __attribute__((ext_vector_type(16))) _Float16 v16h;
typedef __attribute__((ext_vector_type(8)))  float    v8f;

constexpr int N_NODES = 50000;
constexpr int E_EDGES = 800000;
constexpr int D       = 96;              // D_IN == D_HID == 96
constexpr int KTILES  = D / 32;          // 3
constexpr int NTILES  = D / 16;          // 6
constexpr int MTILES  = N_NODES / 16;    // 3125 (exact)

// packed weight fragment: [kt][nt][lane(32)][j(16)] halfs
constexpr int FRAG_HALFS = 32 * 16;                       // 512
constexpr int PACKW_HALFS = KTILES * NTILES * FRAG_HALFS; // 9216
constexpr size_t PACKW_BYTES = PACKW_HALFS * sizeof(_Float16); // 18432

__global__ void zero_f32(float* __restrict__ p, int n) {
  int i = blockIdx.x * blockDim.x + threadIdx.x;
  if (i < n) p[i] = 0.0f;
}

__global__ void init_out(float* __restrict__ out, const float* __restrict__ b_lin, int n) {
  int i = blockIdx.x * blockDim.x + threadIdx.x;
  if (i < n) out[i] = b_lin[0];
}

// Pre-swizzle a [96,96] row-major f32 weight (k rows, n cols) into WMMA
// B-fragment layout: lane l+16h holds column n=l, k = 8*h + j (j<8), 16 + 8*h + (j-8) (j>=8).
__global__ void pack_w(const float* __restrict__ w, _Float16* __restrict__ pk) {
  int tid = blockIdx.x * blockDim.x + threadIdx.x;    // 576 threads
  if (tid >= KTILES * NTILES * 32) return;
  int lane = tid & 31;
  int nt   = (tid >> 5) % NTILES;
  int kt   = tid / (32 * NTILES);
  int n    = nt * 16 + (lane & 15);
  int kh   = 8 * (lane >> 4);
#pragma unroll
  for (int j = 0; j < 16; ++j) {
    int k = kt * 32 + kh + j + (j >= 8 ? 8 : 0);
    pk[((kt * NTILES + nt) * 32 + lane) * 16 + j] = (_Float16)w[k * D + n];
  }
}

__global__ void deg_count(const int* __restrict__ dst, float* __restrict__ deg, int e) {
  int i = blockIdx.x * blockDim.x + threadIdx.x;
  if (i < e) atomicAdd(&deg[dst[i]], 1.0f);
}

// agg[dst[e], f] += feat[src[e], f]   over all (e, f)
__global__ void scatter_add(const int* __restrict__ src, const int* __restrict__ dst,
                            const float* __restrict__ feat, float* __restrict__ agg,
                            int total) {
  int i = blockIdx.x * blockDim.x + threadIdx.x;
  if (i >= total) return;
  int e = i / D;
  int f = i - e * D;
  atomicAdd(&agg[dst[e] * D + f], feat[src[e] * D + f]);
}

// One wave per 16x16 output tile:
//   C = (agg*invdeg) @ Wl + xin @ Wr   (f16 WMMA, f32 acc), + bias, ReLU.
// Two independent accumulators (c1, c2) so consecutive WMMAs form two
// independent dependency chains (hides the 5-cycle f16 WMMA RAW hazard).
// FUSE_OUT=0: store h tile.  FUSE_OUT=1: atomicAdd(out[m], relu(c)*w_lin[n]).
template <int FUSE_OUT>
__global__ __launch_bounds__(32) void sage_gemm(
    const float* __restrict__ agg, const float* __restrict__ deg,
    const float* __restrict__ xin,
    const _Float16* __restrict__ pWl, const _Float16* __restrict__ pWr,
    const float* __restrict__ bias,
    float* __restrict__ hout,          // FUSE_OUT==0
    const float* __restrict__ wlin,    // FUSE_OUT==1
    float* __restrict__ logits) {      // FUSE_OUT==1
  const int mTile = blockIdx.x / NTILES;
  const int nTile = blockIdx.x % NTILES;
  const int lane  = threadIdx.x;
  const int l     = lane & 15;
  const int h     = lane >> 4;

  const int m = mTile * 16 + l;                 // A row for this lane
  // v_rcp_f32: ~1 ulp, plenty for an f16-bound operand; avoids the full
  // IEEE div_scale/div_fmas Newton-Raphson sequence in the tile prologue.
  const float invd = __builtin_amdgcn_rcpf(fmaxf(deg[m], 1.0f));

  v8f c1 = {};                                  // chain 1: (agg*invd) @ Wl
  v8f c2 = {};                                  // chain 2: xin @ Wr
#pragma unroll
  for (int kt = 0; kt < KTILES; ++kt) {
    const int kbase = kt * 32 + 8 * h;          // lane's two contiguous k-runs
    {
      const float* rp = agg + (size_t)m * D + kbase;
      v16h a;
#pragma unroll
      for (int j = 0; j < 8; ++j) a[j] = (_Float16)(rp[j] * invd);
#pragma unroll
      for (int j = 0; j < 8; ++j) a[8 + j] = (_Float16)(rp[16 + j] * invd);
      v16h b = *(const v16h*)(pWl + ((kt * NTILES + nTile) * 32 + lane) * 16);
      c1 = __builtin_amdgcn_wmma_f32_16x16x32_f16(false, a, false, b,
                                                  (short)0, c1, false, false);
    }
    {
      const float* rp = xin + (size_t)m * D + kbase;
      v16h a;
#pragma unroll
      for (int j = 0; j < 8; ++j) a[j] = (_Float16)rp[j];
#pragma unroll
      for (int j = 0; j < 8; ++j) a[8 + j] = (_Float16)rp[16 + j];
      v16h b = *(const v16h*)(pWr + ((kt * NTILES + nTile) * 32 + lane) * 16);
      c2 = __builtin_amdgcn_wmma_f32_16x16x32_f16(false, a, false, b,
                                                  (short)0, c2, false, false);
    }
  }

  const int n = nTile * 16 + l;                 // C/D column for this lane
  const float bv = bias[n];
  if (FUSE_OUT) {
    const float wl = wlin[n];
#pragma unroll
    for (int r = 0; r < 8; ++r) {
      int mo = mTile * 16 + r + 8 * h;
      float v = fmaxf(c1[r] + c2[r] + bv, 0.0f);
      atomicAdd(&logits[mo], v * wl);
    }
  } else {
#pragma unroll
    for (int r = 0; r < 8; ++r) {
      int mo = mTile * 16 + r + 8 * h;
      hout[(size_t)mo * D + n] = fmaxf(c1[r] + c2[r] + bv, 0.0f);
    }
  }
}

extern "C" void kernel_launch(void* const* d_in, const int* in_sizes, int n_in,
                              void* d_out, int out_size, void* d_ws, size_t ws_size,
                              hipStream_t stream) {
  const float* x    = (const float*)d_in[0];
  const int*   ei   = (const int*)d_in[1];          // [2,E] int32
  const float* w1_l = (const float*)d_in[2];
  const float* b1   = (const float*)d_in[3];
  const float* w1_r = (const float*)d_in[4];
  const float* w2_l = (const float*)d_in[5];
  const float* b2   = (const float*)d_in[6];
  const float* w2_r = (const float*)d_in[7];
  const float* wlin = (const float*)d_in[8];
  const float* blin = (const float*)d_in[9];
  float* out = (float*)d_out;

  const int* src = ei;
  const int* dst = ei + E_EDGES;

  // workspace layout (256B aligned regions)
  char* ws = (char*)d_ws;
  auto align256 = [](size_t v) { return (v + 255) & ~(size_t)255; };
  size_t off = 0;
  float* deg = (float*)(ws + off); off = align256(off + (size_t)N_NODES * 4);
  float* agg = (float*)(ws + off); off = align256(off + (size_t)N_NODES * D * 4);
  float* h1  = (float*)(ws + off); off = align256(off + (size_t)N_NODES * D * 4);
  _Float16* pw1l = (_Float16*)(ws + off); off = align256(off + PACKW_BYTES);
  _Float16* pw1r = (_Float16*)(ws + off); off = align256(off + PACKW_BYTES);
  _Float16* pw2l = (_Float16*)(ws + off); off = align256(off + PACKW_BYTES);
  _Float16* pw2r = (_Float16*)(ws + off); off = align256(off + PACKW_BYTES);

  const int featN   = N_NODES * D;                  // 4,800,000
  const int scatTot = E_EDGES * D;                  // 76,800,000

  // 0) zero accumulators, pack weights into WMMA B-fragment layout
  zero_f32<<<(N_NODES + 255) / 256, 256, 0, stream>>>(deg, N_NODES);
  zero_f32<<<(featN + 255) / 256, 256, 0, stream>>>(agg, featN);
  pack_w<<<18, 32, 0, stream>>>(w1_l, pw1l);
  pack_w<<<18, 32, 0, stream>>>(w1_r, pw1r);
  pack_w<<<18, 32, 0, stream>>>(w2_l, pw2l);
  pack_w<<<18, 32, 0, stream>>>(w2_r, pw2r);

  // 1) degrees (shared by both layers)
  deg_count<<<(E_EDGES + 255) / 256, 256, 0, stream>>>(dst, deg, E_EDGES);

  // 2) layer 1 aggregation + fused SAGE linear (WMMA) -> h1
  scatter_add<<<(scatTot + 255) / 256, 256, 0, stream>>>(src, dst, x, agg, scatTot);
  sage_gemm<0><<<MTILES * NTILES, 32, 0, stream>>>(agg, deg, x, pw1l, pw1r, b1,
                                                   h1, nullptr, nullptr);

  // 3) layer 2 aggregation + fused SAGE linear + fused readout -> logits
  zero_f32<<<(featN + 255) / 256, 256, 0, stream>>>(agg, featN);
  scatter_add<<<(scatTot + 255) / 256, 256, 0, stream>>>(src, dst, h1, agg, scatTot);
  init_out<<<(N_NODES + 255) / 256, 256, 0, stream>>>(out, blin, N_NODES);
  sage_gemm<1><<<MTILES * NTILES, 32, 0, stream>>>(agg, deg, h1, pw2l, pw2r, b2,
                                                   nullptr, wlin, out);
}
